// DualBranchEEGModel_78623671320902
// MI455X (gfx1250) — compile-verified, matrix operand order checked by reading, not run on hardware
//
#include <hip/hip_runtime.h>
#include <cstdint>
#include <cstddef>

// ---------------------------------------------------------------------------
// Model dims
// ---------------------------------------------------------------------------
#define TT   200      // T
#define NSEQ 2048     // B*C
#define HH   64       // H
#define BB   16
#define CC   128
#define EE   262144   // edges

typedef __attribute__((ext_vector_type(16))) _Float16 v16h;
typedef __attribute__((ext_vector_type(8)))  float    v8f;

// ---------------------------------------------------------------------------
// Fast math (hardware TRANS ops instead of IEEE div/sqrt chains)
// ---------------------------------------------------------------------------
__device__ __forceinline__ float fast_rcp(float x) { return __builtin_amdgcn_rcpf(x); }
__device__ __forceinline__ float fast_rsq(float x) { return __builtin_amdgcn_rsqf(x); }
__device__ __forceinline__ float sigmoidf_(float x) { return fast_rcp(1.0f + __expf(-x)); }
__device__ __forceinline__ float tanhf_(float x) {
  // tanh(x) = 1 - 2 / (1 + exp(2x))
  return 1.0f - 2.0f * fast_rcp(1.0f + __expf(2.0f * x));
}

// ---------------------------------------------------------------------------
// WMMA fragment helpers (CDNA5 16-bit A/B layout, ISA 7.12.2)
//   lane 0..15 : rows M (or N), k-half 0 ; lane 16..31: same rows, k-half 1
//   VGPR v<4 : K = 2v,2v+1 (+8 for half1); v>=4: K = 16+2(v-4).. (+8)
// ---------------------------------------------------------------------------
__device__ __forceinline__ constexpr int frag_k(int e, int kh) {
  int v = e >> 1;
  int base = (v < 4) ? (2 * v) : (16 + 2 * (v - 4));
  return base + (e & 1) + (kh ? 8 : 0);
}

// K and LDK are compile-time so all bounds checks fold away (no branches, no
// guarded loads -> loads batch up and the compiler emits one wait per block).
template <int K, int LDK, typename SRC>
__device__ __forceinline__ v16h frag_ld(const SRC* base, int k0, int kh) {
  v16h f;
#pragma unroll
  for (int e = 0; e < 16; ++e) {
    int k = k0 + frag_k(e, kh);
    f[e] = (k < K) ? (_Float16)base[(size_t)k * LDK] : (_Float16)0.0f;
  }
  return f;
}

__device__ __forceinline__ v8f wmma16(v16h a, v16h b, v8f c) {
  // D = A(16x32 f16) * B(32x16 f16) + C(16x16 f32)
  return __builtin_amdgcn_wmma_f32_16x16x32_f16(false, a, false, b, (short)0, c,
                                                false, false);
}

// ---------------------------------------------------------------------------
// Generic WMMA GEMM:  out[m, coloff+n] = act( A(MxK) * B^T + bias )
//   B element (n,k) at Bw[n*ldn + k*LDK]  (handles both (N,K) and (K,N) weights)
//   one wave per 16x16 output tile; M,N multiples of 16; K zero-padded at
//   compile time (template).
// ---------------------------------------------------------------------------
template <int K, int LDK>
__global__ void __launch_bounds__(32)
gemm_wmma_kernel(const float* __restrict__ A, int lda,
                 const float* __restrict__ Bw, int ldn,
                 const float* __restrict__ bias,
                 float* __restrict__ Cc, int ldc, int coloff,
                 int relu) {
  int lane = threadIdx.x;
  int ln = lane & 15;
  int kh = lane >> 4;
  int m0 = blockIdx.x * 16;
  int n0 = blockIdx.y * 16;
  const float* arow = A + (size_t)(m0 + ln) * lda;
  const float* brow = Bw + (size_t)(n0 + ln) * ldn;
  v8f acc = {};
#pragma unroll
  for (int k0 = 0; k0 < K; k0 += 32) {
    v16h a = frag_ld<K, 1>(arow, k0, kh);
    v16h b = frag_ld<K, LDK>(brow, k0, kh);
    acc = wmma16(a, b, acc);
  }
  int n = n0 + ln;
  float bv = bias ? bias[n] : 0.0f;
#pragma unroll
  for (int v = 0; v < 8; ++v) {
    int m = m0 + v + (kh ? 8 : 0);
    float val = acc[v] + bv;
    if (relu) val = fmaxf(val, 0.0f);
    Cc[(size_t)m * ldc + coloff + n] = val;
  }
}

// ---------------------------------------------------------------------------
// GRU layer 0 (input dim 1). One wave handles 16 sequences, full 200-step scan.
// x tile and Whh live in LDS; recurrent GEMM h(16x64) @ Whh^T(64x192) via WMMA.
// Writes hidden state sequence (f16) into l1[seq][t][dir*64 + k].
// ---------------------------------------------------------------------------
__global__ void __launch_bounds__(32)
gru_layer0_kernel(const float* __restrict__ x,      // (2048, 200)
                  const float* __restrict__ Wih0,   // (2, 192, 1)
                  const float* __restrict__ Whh0,   // (2, 192, 64)
                  const float* __restrict__ b0,     // (2, 2, 192)
                  _Float16* __restrict__ l1)        // (2048, 200, 128)
{
  __shared__ _Float16 whh[192 * 64];   // 24.0 KB
  __shared__ float    xs[16 * TT];     // 12.5 KB
  __shared__ float    hbuf[16 * 64];   //  4.0 KB
  int lane = threadIdx.x;
  int dir  = blockIdx.y;
  int row0 = blockIdx.x * 16;

  const float* W = Whh0 + (size_t)dir * 192 * 64;
  for (int i = lane; i < 192 * 64; i += 32) whh[i] = (_Float16)W[i];
  for (int i = lane; i < 16 * TT; i += 32) {
    int r = i / TT, t = i % TT;
    xs[i] = x[(size_t)(row0 + r) * TT + t];
  }
  for (int i = lane; i < 16 * 64; i += 32) hbuf[i] = 0.0f;
  __syncthreads();

  int ln = lane & 15;
  int kh = lane >> 4;
  const float* wih = Wih0 + (size_t)dir * 192;
  const float* bih = b0 + (size_t)dir * 2 * 192;
  const float* bhh = bih + 192;
  float wr[4], wz[4], wn[4], bir[4], biz[4], bin_[4], bhr[4], bhz[4], bhn[4];
#pragma unroll
  for (int j = 0; j < 4; ++j) {
    int n = j * 16 + ln;
    wr[j] = wih[n];        wz[j] = wih[64 + n];   wn[j] = wih[128 + n];
    bir[j] = bih[n];       biz[j] = bih[64 + n];  bin_[j] = bih[128 + n];
    bhr[j] = bhh[n];       bhz[j] = bhh[64 + n];  bhn[j] = bhh[128 + n];
  }

  for (int step = 0; step < TT; ++step) {
    int t = dir ? (TT - 1 - step) : step;
    const float* hrow = hbuf + ln * 64;
    v16h a0 = frag_ld<64, 1>(hrow, 0, kh);
    v16h a1 = frag_ld<64, 1>(hrow, 32, kh);
#pragma unroll
    for (int j = 0; j < 4; ++j) {
      const _Float16* br = whh + (size_t)(j * 16 + ln) * 64;
      const _Float16* bz = whh + (size_t)(64 + j * 16 + ln) * 64;
      const _Float16* bn = whh + (size_t)(128 + j * 16 + ln) * 64;
      v8f accR = {}, accZ = {}, accN = {};
      accR = wmma16(a0, frag_ld<64, 1>(br, 0, kh), accR);
      accR = wmma16(a1, frag_ld<64, 1>(br, 32, kh), accR);
      accZ = wmma16(a0, frag_ld<64, 1>(bz, 0, kh), accZ);
      accZ = wmma16(a1, frag_ld<64, 1>(bz, 32, kh), accZ);
      accN = wmma16(a0, frag_ld<64, 1>(bn, 0, kh), accN);
      accN = wmma16(a1, frag_ld<64, 1>(bn, 32, kh), accN);
#pragma unroll
      for (int v = 0; v < 8; ++v) {
        int m = v + (kh ? 8 : 0);
        float xv = xs[m * TT + t];
        float r  = sigmoidf_(xv * wr[j] + bir[j] + accR[v] + bhr[j]);
        float z  = sigmoidf_(xv * wz[j] + biz[j] + accZ[v] + bhz[j]);
        float hn = accN[v] + bhn[j];
        float nn = tanhf_(xv * wn[j] + bin_[j] + r * hn);
        int col = j * 16 + ln;
        float hp = hbuf[m * 64 + col];
        float hnew = (1.0f - z) * nn + z * hp;
        hbuf[m * 64 + col] = hnew;
        l1[((size_t)(row0 + m) * TT + t) * 128 + dir * 64 + col] = (_Float16)hnew;
      }
    }
    __syncthreads();
  }
}

// ---------------------------------------------------------------------------
// GRU layer 1 (input dim 128 = concat fwd/bwd of layer 0).
// Input-gate GEMM (16x128 @ 128x192) and recurrent GEMM (16x64 @ 64x192) both
// via WMMA. Only l2[:, -1] is needed: fwd runs 200 steps, bwd needs exactly
// ONE step (t=199), since reversed-scan output at original t=199 is the state
// after its first step from h0=0.
// ---------------------------------------------------------------------------
__global__ void __launch_bounds__(32)
gru_layer1_kernel(const _Float16* __restrict__ l1,   // (2048, 200, 128)
                  const float* __restrict__ Wih1,    // (2, 192, 128)
                  const float* __restrict__ Whh1,    // (2, 192, 64)
                  const float* __restrict__ b1,      // (2, 2, 192)
                  float* __restrict__ gru_last)      // (2048, 128)
{
  __shared__ _Float16 wih[192 * 128];  // 48 KB
  __shared__ _Float16 whh[192 * 64];   // 24 KB
  __shared__ float    hbuf[16 * 64];   //  4 KB
  int lane = threadIdx.x;
  int dir  = blockIdx.y;
  int row0 = blockIdx.x * 16;

  const float* Wi = Wih1 + (size_t)dir * 192 * 128;
  const float* Wh = Whh1 + (size_t)dir * 192 * 64;
  for (int i = lane; i < 192 * 128; i += 32) wih[i] = (_Float16)Wi[i];
  for (int i = lane; i < 192 * 64; i += 32) whh[i] = (_Float16)Wh[i];
  for (int i = lane; i < 16 * 64; i += 32) hbuf[i] = 0.0f;
  __syncthreads();

  int ln = lane & 15;
  int kh = lane >> 4;
  const float* bih = b1 + (size_t)dir * 2 * 192;
  const float* bhh = bih + 192;
  float br_[4], bz_[4], bin_[4], bhn_[4];
#pragma unroll
  for (int j = 0; j < 4; ++j) {
    int n = j * 16 + ln;
    br_[j]  = bih[n] + bhh[n];
    bz_[j]  = bih[64 + n] + bhh[64 + n];
    bin_[j] = bih[128 + n];
    bhn_[j] = bhh[128 + n];
  }

  int nsteps = dir ? 1 : TT;
  for (int step = 0; step < nsteps; ++step) {
    int t = dir ? (TT - 1) : step;
    // input-sequence fragments (K = 128)
    const _Float16* xrow = l1 + ((size_t)(row0 + ln) * TT + t) * 128;
    v16h ax[4];
#pragma unroll
    for (int kf = 0; kf < 4; ++kf) ax[kf] = frag_ld<128, 1>(xrow, kf * 32, kh);
    const float* hrow = hbuf + ln * 64;
    v16h ah0 = frag_ld<64, 1>(hrow, 0, kh);
    v16h ah1 = frag_ld<64, 1>(hrow, 32, kh);
#pragma unroll
    for (int j = 0; j < 4; ++j) {
      const _Float16* wiR = wih + (size_t)(j * 16 + ln) * 128;
      const _Float16* wiZ = wih + (size_t)(64 + j * 16 + ln) * 128;
      const _Float16* wiN = wih + (size_t)(128 + j * 16 + ln) * 128;
      const _Float16* whR = whh + (size_t)(j * 16 + ln) * 64;
      const _Float16* whZ = whh + (size_t)(64 + j * 16 + ln) * 64;
      const _Float16* whN = whh + (size_t)(128 + j * 16 + ln) * 64;
      v8f accR = {}, accZ = {}, accXN = {}, accHN = {};
#pragma unroll
      for (int kf = 0; kf < 4; ++kf) {
        accR  = wmma16(ax[kf], frag_ld<128, 1>(wiR, kf * 32, kh), accR);
        accZ  = wmma16(ax[kf], frag_ld<128, 1>(wiZ, kf * 32, kh), accZ);
        accXN = wmma16(ax[kf], frag_ld<128, 1>(wiN, kf * 32, kh), accXN);
      }
      accR  = wmma16(ah0, frag_ld<64, 1>(whR, 0, kh), accR);
      accR  = wmma16(ah1, frag_ld<64, 1>(whR, 32, kh), accR);
      accZ  = wmma16(ah0, frag_ld<64, 1>(whZ, 0, kh), accZ);
      accZ  = wmma16(ah1, frag_ld<64, 1>(whZ, 32, kh), accZ);
      accHN = wmma16(ah0, frag_ld<64, 1>(whN, 0, kh), accHN);
      accHN = wmma16(ah1, frag_ld<64, 1>(whN, 32, kh), accHN);
#pragma unroll
      for (int v = 0; v < 8; ++v) {
        int m = v + (kh ? 8 : 0);
        float r  = sigmoidf_(accR[v] + br_[j]);
        float z  = sigmoidf_(accZ[v] + bz_[j]);
        float nn = tanhf_(accXN[v] + bin_[j] + r * (accHN[v] + bhn_[j]));
        int col = j * 16 + ln;
        float hp = hbuf[m * 64 + col];
        float hnew = (1.0f - z) * nn + z * hp;
        hbuf[m * 64 + col] = hnew;
        if (t == TT - 1)
          gru_last[(size_t)(row0 + m) * 128 + dir * 64 + col] = hnew;
      }
    }
    __syncthreads();
  }
}

// ---------------------------------------------------------------------------
// GCN helpers
// ---------------------------------------------------------------------------
__global__ void zero_f32_kernel(float* p, int n) {
  int i = blockIdx.x * blockDim.x + threadIdx.x;
  if (i < n) p[i] = 0.0f;
}

__global__ void deg_kernel(const int* __restrict__ dst, const float* __restrict__ w,
                           float* __restrict__ deg, int E, int N) {
  int i = blockIdx.x * blockDim.x + threadIdx.x;
  if (i < E) atomicAdd(&deg[dst[i]], w[i]);
  else if (i < E + N) atomicAdd(&deg[i - E], 1.0f);   // self loop weight 1
}

__global__ void dinv_kernel(const float* __restrict__ deg, float* __restrict__ dinv, int N) {
  int i = blockIdx.x * blockDim.x + threadIdx.x;
  if (i < N) dinv[i] = (deg[i] > 0.0f) ? fast_rsq(deg[i]) : 0.0f;
}

// acc[i][f] = hW[i][f] * dinv[i]^2 (self loop) + bias[f]
__global__ void gcn_init_kernel(const float* __restrict__ hW, const float* __restrict__ dinv,
                                const float* __restrict__ bias, float* __restrict__ acc, int N) {
  int idx = blockIdx.x * blockDim.x + threadIdx.x;
  int i = idx >> 6, f = idx & 63;
  if (i < N) {
    float di = dinv[i];
    acc[(size_t)i * 64 + f] = hW[(size_t)i * 64 + f] * di * di + bias[f];
  }
}

// one thread per (edge, feature): acc[dst] += hW[src] * dinv[src]*w*dinv[dst]
__global__ void gcn_scatter_kernel(const int* __restrict__ src, const int* __restrict__ dst,
                                   const float* __restrict__ w, const float* __restrict__ dinv,
                                   const float* __restrict__ hW, float* __restrict__ acc, int E) {
  int tid = blockIdx.x * blockDim.x + threadIdx.x;
  int e = tid >> 6, f = tid & 63;
  if (e >= E) return;
  int s = src[e], d = dst[e];
  float norm = dinv[s] * w[e] * dinv[d];
  atomicAdd(&acc[(size_t)d * 64 + f], hW[(size_t)s * 64 + f] * norm);
}

// per-feature mean / population var over N rows (blockIdx.x = feature)
__global__ void bn_stats_kernel(const float* __restrict__ h, float* __restrict__ mu,
                                float* __restrict__ var, int N) {
  int f = blockIdx.x;
  float s = 0.0f, s2 = 0.0f;
  for (int i = threadIdx.x; i < N; i += 256) {
    float v = h[(size_t)i * 64 + f];
    s += v; s2 += v * v;
  }
  __shared__ float ss[256], ss2[256];
  ss[threadIdx.x] = s; ss2[threadIdx.x] = s2;
  __syncthreads();
  for (int off = 128; off > 0; off >>= 1) {
    if (threadIdx.x < off) { ss[threadIdx.x] += ss[threadIdx.x + off]; ss2[threadIdx.x] += ss2[threadIdx.x + off]; }
    __syncthreads();
  }
  if (threadIdx.x == 0) {
    float m = ss[0] / (float)N;
    mu[f] = m;
    var[f] = ss2[0] / (float)N - m * m;
  }
}

__global__ void bn_apply_relu_kernel(const float* __restrict__ h, const float* __restrict__ g,
                                     const float* __restrict__ b, const float* __restrict__ mu,
                                     const float* __restrict__ var, float* __restrict__ out,
                                     int ldc, int coloff, int N) {
  int idx = blockIdx.x * blockDim.x + threadIdx.x;
  int i = idx >> 6, f = idx & 63;
  if (i >= N) return;
  float v = g[f] * (h[(size_t)i * 64 + f] - mu[f]) * fast_rsq(var[f] + 1e-5f) + b[f];
  out[(size_t)i * ldc + coloff + f] = fmaxf(v, 0.0f);
}

// ---------------------------------------------------------------------------
// Fusion tail
// ---------------------------------------------------------------------------
__global__ void pool_kernel(const float* __restrict__ fused2, float* __restrict__ pooled) {
  int tid = blockIdx.x * blockDim.x + threadIdx.x;   // 16*128 threads
  if (tid >= BB * 128) return;
  int b = tid >> 7, f = tid & 127;
  float s = 0.0f;
  for (int c = 0; c < CC; ++c) s += fused2[((size_t)(b * CC + c)) * 128 + f];
  pooled[b * 128 + f] = s * (1.0f / (float)CC);
}

__global__ void pred2_kernel(const float* __restrict__ hmid, const float* __restrict__ W,
                             const float* __restrict__ b, float* __restrict__ out) {
  int i = threadIdx.x;
  if (i >= BB) return;
  float s = b[0];
  for (int k = 0; k < 64; ++k) s += hmid[(size_t)i * 64 + k] * W[k];
  out[i] = s;
}

// ---------------------------------------------------------------------------
// Host-side launch
// ---------------------------------------------------------------------------
static inline size_t align256(size_t x) { return (x + 255) & ~(size_t)255; }

extern "C" void kernel_launch(void* const* d_in, const int* in_sizes, int n_in,
                              void* d_out, int out_size, void* d_ws, size_t ws_size,
                              hipStream_t stream) {
  (void)in_sizes; (void)n_in; (void)out_size; (void)ws_size;
  const float* x        = (const float*)d_in[0];    // (16,128,200) == (2048,200)
  const int*   eidx     = (const int*)d_in[1];      // (2, E)
  const float* ew       = (const float*)d_in[2];    // (E,)
  const float* gru_Wih0 = (const float*)d_in[3];
  const float* gru_Whh0 = (const float*)d_in[4];
  const float* gru_b0   = (const float*)d_in[5];
  const float* gru_Wih1 = (const float*)d_in[6];
  const float* gru_Whh1 = (const float*)d_in[7];
  const float* gru_b1   = (const float*)d_in[8];
  const float* proj_W   = (const float*)d_in[9];
  const float* proj_b   = (const float*)d_in[10];
  const float* gcn_W1   = (const float*)d_in[11];
  const float* gcn_b1   = (const float*)d_in[12];
  const float* gcn_W2   = (const float*)d_in[13];
  const float* gcn_b2   = (const float*)d_in[14];
  const float* bn_g1    = (const float*)d_in[15];
  const float* bn_b1    = (const float*)d_in[16];
  const float* bn_g2    = (const float*)d_in[17];
  const float* bn_b2    = (const float*)d_in[18];
  const float* fus_W1   = (const float*)d_in[19];
  const float* fus_b1   = (const float*)d_in[20];
  const float* fus_W2   = (const float*)d_in[21];
  const float* fus_b2   = (const float*)d_in[22];
  const float* pred_W1  = (const float*)d_in[23];
  const float* pred_b1  = (const float*)d_in[24];
  const float* pred_W2  = (const float*)d_in[25];
  const float* pred_b2  = (const float*)d_in[26];
  float* out = (float*)d_out;

  // workspace carve-up
  char* w = (char*)d_ws;
  size_t off = 0;
  _Float16* l1    = (_Float16*)(w + off); off = align256(off + (size_t)NSEQ * TT * 128 * 2);
  float* gru_last = (float*)(w + off);    off = align256(off + (size_t)NSEQ * 128 * 4);
  float* hW       = (float*)(w + off);    off = align256(off + (size_t)NSEQ * 64 * 4);
  float* acc      = (float*)(w + off);    off = align256(off + (size_t)NSEQ * 64 * 4);
  float* g1       = (float*)(w + off);    off = align256(off + (size_t)NSEQ * 64 * 4);
  float* deg      = (float*)(w + off);    off = align256(off + (size_t)NSEQ * 4);
  float* dinv     = (float*)(w + off);    off = align256(off + (size_t)NSEQ * 4);
  float* mu1      = (float*)(w + off);    off = align256(off + 64 * 4);
  float* var1     = (float*)(w + off);    off = align256(off + 64 * 4);
  float* mu2      = (float*)(w + off);    off = align256(off + 64 * 4);
  float* var2     = (float*)(w + off);    off = align256(off + 64 * 4);
  float* fused    = (float*)(w + off);    off = align256(off + (size_t)NSEQ * 128 * 4);
  float* fus1o    = (float*)(w + off);    off = align256(off + (size_t)NSEQ * 128 * 4);
  float* fus2o    = (float*)(w + off);    off = align256(off + (size_t)NSEQ * 128 * 4);
  float* pooled   = (float*)(w + off);    off = align256(off + (size_t)BB * 128 * 4);
  float* hmid     = (float*)(w + off);    off = align256(off + (size_t)BB * 64 * 4);

  const int* esrc = eidx;
  const int* edst = eidx + EE;

  // ---- GRU branch ----
  gru_layer0_kernel<<<dim3(NSEQ / 16, 2), 32, 0, stream>>>(x, gru_Wih0, gru_Whh0, gru_b0, l1);
  gru_layer1_kernel<<<dim3(NSEQ / 16, 2), 32, 0, stream>>>(l1, gru_Wih1, gru_Whh1, gru_b1, gru_last);
  // gru_feat = gru_last @ proj_W^T + proj_b  -> fused[:, 0:64]
  gemm_wmma_kernel<128, 1><<<dim3(NSEQ / 16, 64 / 16), 32, 0, stream>>>(
      gru_last, 128, proj_W, 128, proj_b, fused, 128, 0, 0);

  // ---- GCN branch ----
  zero_f32_kernel<<<(NSEQ + 255) / 256, 256, 0, stream>>>(deg, NSEQ);
  deg_kernel<<<(EE + NSEQ + 255) / 256, 256, 0, stream>>>(edst, ew, deg, EE, NSEQ);
  dinv_kernel<<<(NSEQ + 255) / 256, 256, 0, stream>>>(deg, dinv, NSEQ);

  // layer 1: hW = node(2048x200) @ gcn_W1(200x64)   (W is (K,N): ldn=1, LDK=64)
  gemm_wmma_kernel<200, 64><<<dim3(NSEQ / 16, 64 / 16), 32, 0, stream>>>(
      x, TT, gcn_W1, 1, nullptr, hW, 64, 0, 0);
  gcn_init_kernel<<<(NSEQ * 64) / 256, 256, 0, stream>>>(hW, dinv, gcn_b1, acc, NSEQ);
  gcn_scatter_kernel<<<(size_t)EE * 64 / 256, 256, 0, stream>>>(esrc, edst, ew, dinv, hW, acc, EE);
  bn_stats_kernel<<<64, 256, 0, stream>>>(acc, mu1, var1, NSEQ);
  bn_apply_relu_kernel<<<(NSEQ * 64) / 256, 256, 0, stream>>>(acc, bn_g1, bn_b1, mu1, var1, g1, 64, 0, NSEQ);

  // layer 2: hW = g1(2048x64) @ gcn_W2(64x64)
  gemm_wmma_kernel<64, 64><<<dim3(NSEQ / 16, 64 / 16), 32, 0, stream>>>(
      g1, 64, gcn_W2, 1, nullptr, hW, 64, 0, 0);
  gcn_init_kernel<<<(NSEQ * 64) / 256, 256, 0, stream>>>(hW, dinv, gcn_b2, acc, NSEQ);
  gcn_scatter_kernel<<<(size_t)EE * 64 / 256, 256, 0, stream>>>(esrc, edst, ew, dinv, hW, acc, EE);
  bn_stats_kernel<<<64, 256, 0, stream>>>(acc, mu2, var2, NSEQ);
  // gnn_feat -> fused[:, 64:128]
  bn_apply_relu_kernel<<<(NSEQ * 64) / 256, 256, 0, stream>>>(acc, bn_g2, bn_b2, mu2, var2, fused, 128, 64, NSEQ);

  // ---- fusion + prediction ----
  gemm_wmma_kernel<128, 1><<<dim3(NSEQ / 16, 128 / 16), 32, 0, stream>>>(
      fused, 128, fus_W1, 128, fus_b1, fus1o, 128, 0, 1);
  gemm_wmma_kernel<128, 1><<<dim3(NSEQ / 16, 128 / 16), 32, 0, stream>>>(
      fus1o, 128, fus_W2, 128, fus_b2, fus2o, 128, 0, 1);
  pool_kernel<<<(BB * 128 + 255) / 256, 256, 0, stream>>>(fus2o, pooled);
  gemm_wmma_kernel<128, 1><<<dim3(1, 64 / 16), 32, 0, stream>>>(
      pooled, 128, pred_W1, 128, pred_b1, hmid, 64, 0, 1);
  pred2_kernel<<<1, 32, 0, stream>>>(hmid, pred_W2, pred_b2, out);
}
